// Model_39676907886764
// MI455X (gfx1250) — matrix operand order repaired and checked by reading, not compiled, not run on hardware
//
#include <hip/hip_runtime.h>

#define Bsz 8
#define LQn 2048
#define LKVn 2048
#define Dm 1024

typedef _Float16 half_t;
typedef _Float16 v16h __attribute__((ext_vector_type(16)));
typedef float    v8f  __attribute__((ext_vector_type(8)));

// ---------------- f32 -> f16 weight convert ----------------
__global__ void cvt_f32_f16(const float* __restrict__ src, half_t* __restrict__ dst, int n) {
    int i = blockIdx.x * blockDim.x + threadIdx.x;
    if (i < n) dst[i] = (half_t)src[i];
}

// ---------------- V transpose: f32 [b][kv][d] -> f16 [b][d][kv] ----------------
__global__ void transpose_v(const float* __restrict__ v, half_t* __restrict__ vt) {
    __shared__ half_t t[32][33];
    int b   = blockIdx.z;
    int kv0 = blockIdx.x * 32;
    int d0  = blockIdx.y * 32;
    int tx = threadIdx.x, ty = threadIdx.y;
    t[ty][tx] = (half_t)v[((size_t)b * LKVn + kv0 + ty) * Dm + d0 + tx];
    __syncthreads();
    vt[((size_t)b * Dm + d0 + ty) * LKVn + kv0 + tx] = t[tx][ty];
}

// ---------------- Projection: Y = X @ W^T + bias  (f32 in, f16 out) ----------------
// grid: (Dm/128, M/128), block 256 (8 waves). Double-buffered LDS X staging.
__global__ __launch_bounds__(256) void proj_gemm(
        const float* __restrict__ X, const half_t* __restrict__ W16,
        const float* __restrict__ bias, half_t* __restrict__ Y) {
    __shared__ half_t lds_x[2][128 * 40];        // 128 rows x 32 halves, stride 40
    int tid  = threadIdx.x;
    int lane = tid & 31;
    int wv   = tid >> 5;
    int col  = lane & 15;
    int off8  = (lane >> 4) * 8;
    int off16 = (lane >> 4) * 16;
    int n0 = blockIdx.x * 128;
    int m0 = blockIdx.y * 128;

    auto stage = [&](half_t* buf, int k0) {
        int r  = tid >> 3;                       // 0..31
        int c4 = (tid & 7) * 4;
        for (int p = 0; p < 4; ++p) {
            int row = r + p * 32;
            float4 xv = *(const float4*)&X[((size_t)(m0 + row)) * Dm + k0 + c4];
            half_t* d = &buf[row * 40 + c4];
            d[0] = (half_t)xv.x; d[1] = (half_t)xv.y;
            d[2] = (half_t)xv.z; d[3] = (half_t)xv.w;
        }
    };

    v8f acc[8];
    for (int s = 0; s < 8; ++s) acc[s] = (v8f){0,0,0,0,0,0,0,0};

    stage(lds_x[0], 0);
    __syncthreads();
    int cur = 0;
    for (int k0 = 0; k0 < Dm; k0 += 32) {
        if (k0 + 32 < Dm) stage(lds_x[cur ^ 1], k0 + 32);   // overlaps with WMMAs below
        v16h a;
        const half_t* ab = &lds_x[cur][(wv * 16 + col) * 40 + off16];
        *(uint4*)&a       = *(const uint4*)ab;
        *((uint4*)&a + 1) = *(const uint4*)(ab + 8);
        for (int s = 0; s < 8; ++s) {
            const half_t* bb = &W16[(size_t)(n0 + s * 16 + col) * Dm + k0 + off16];
            v16h bf;
            *(uint4*)&bf       = *(const uint4*)bb;
            *((uint4*)&bf + 1) = *(const uint4*)(bb + 8);
            acc[s] = __builtin_amdgcn_wmma_f32_16x16x32_f16(
                         false, a, false, bf, (short)0, acc[s], false, false);
        }
        __syncthreads();
        cur ^= 1;
    }
    int hi8 = (lane >> 4) * 8;
    for (int s = 0; s < 8; ++s) {
        int n = n0 + s * 16 + col;
        float bv = bias[n];
        for (int v = 0; v < 8; ++v) {
            int m = m0 + wv * 16 + v + hi8;
            Y[(size_t)m * Dm + n] = (half_t)(acc[s][v] + bv);
        }
    }
}

// ---------------- Fused flash attention ----------------
// grid: (LQ/32, B), block 512 (16 waves). WG: 32 q-rows.
// Wave w: row-group rg = w>>3 (16 rows), d-chunk / kv-subtile dc = w&7.
__global__ __launch_bounds__(512) void attn_fused(
        const half_t* __restrict__ Q16, const half_t* __restrict__ K16,
        const half_t* __restrict__ VT16, const int* __restrict__ scale_p,
        float* __restrict__ Out) {
    __shared__ half_t lds_q[32 * 1032];          // 32 x 1024 halves, stride 1032
    __shared__ float  lds_s[32 * 132];           // S tile 32x128, stride 132
    __shared__ half_t lds_p[32 * 136];           // P tile 32x128 f16, stride 136
    __shared__ float  lds_red[512];
    __shared__ float  lds_m[32], lds_l[32], lds_f[32];

    int tid  = threadIdx.x;
    int lane = tid & 31;
    int wv   = tid >> 5;                         // 0..15
    int rg   = wv >> 3;                          // row-group 0/1
    int dc   = wv & 7;                           // d-chunk / kv-subtile
    int col  = lane & 15;
    int off8  = (lane >> 4) * 8;
    int off16 = (lane >> 4) * 16;
    int hi8   = off8;
    int q0 = blockIdx.x * 32;
    int b  = blockIdx.y;
    float inv = 1.0f / (float)scale_p[0];

    // stage Q tile (32 x 1024 f16) once
    {
        int r = tid >> 4;                        // 0..31
        int c = (tid & 15) * 64;
        const uint4* src = (const uint4*)&Q16[((size_t)b * LQn + q0 + r) * Dm + c];
        uint4* dst = (uint4*)&lds_q[r * 1032 + c];
        for (int j = 0; j < 8; ++j) dst[j] = src[j];
    }
    if (tid < 32) { lds_m[tid] = -1e30f; lds_l[tid] = 0.0f; }
    __syncthreads();

    v8f oacc[8];
    for (int s = 0; s < 8; ++s) oacc[s] = (v8f){0,0,0,0,0,0,0,0};
    int d0   = dc * 128;
    int arow = rg * 16 + col;                    // A-fragment row (Q and P)

    for (int kv0 = 0; kv0 < LKVn; kv0 += 128) {
        // ---- Phase 1: wave computes S[rg rows][kv0+16*dc ..+16), 2-acc K pipeline ----
        const half_t* krow = &K16[((size_t)b * LKVn + kv0 + dc * 16 + col) * Dm + off16];
        if (kv0 + 128 < LKVn) {                  // warm L2 for next kv-tile
            const char* nk = (const char*)&K16[((size_t)b * LKVn + kv0 + 128 + dc * 16 + col) * Dm];
            __builtin_prefetch(nk, 0, 1);
            __builtin_prefetch(nk + 1024, 0, 1);
            __builtin_prefetch(&VT16[((size_t)b * Dm + d0 + col) * LKVn + kv0 + 128], 0, 1);
        }
        v8f s0 = (v8f){0,0,0,0,0,0,0,0};
        v8f s1 = (v8f){0,0,0,0,0,0,0,0};
        for (int k0 = 0; k0 < Dm; k0 += 64) {
            v16h a, bf;
            const half_t* ab = &lds_q[arow * 1032 + k0 + off16];
            *(uint4*)&a        = *(const uint4*)ab;
            *((uint4*)&a + 1)  = *(const uint4*)(ab + 8);
            const half_t* bb = krow + k0;
            *(uint4*)&bf       = *(const uint4*)bb;
            *((uint4*)&bf + 1) = *(const uint4*)(bb + 8);
            s0 = __builtin_amdgcn_wmma_f32_16x16x32_f16(
                     false, a, false, bf, (short)0, s0, false, false);
            const half_t* ab2 = ab + 32;
            *(uint4*)&a        = *(const uint4*)ab2;
            *((uint4*)&a + 1)  = *(const uint4*)(ab2 + 8);
            const half_t* bb2 = bb + 32;
            *(uint4*)&bf       = *(const uint4*)bb2;
            *((uint4*)&bf + 1) = *(const uint4*)(bb2 + 8);
            s1 = __builtin_amdgcn_wmma_f32_16x16x32_f16(
                     false, a, false, bf, (short)0, s1, false, false);
        }
        v8f sacc = s0 + s1;
        for (int v = 0; v < 8; ++v)
            lds_s[(rg * 16 + v + hi8) * 132 + dc * 16 + col] = sacc[v] * inv;
        __syncthreads();

        // ---- Phase 2a: partial row max (512 threads: row = tid&31, 8 cols each) ----
        {
            int r = tid & 31, g = tid >> 5;
            float pm = -1e30f;
            for (int j = 0; j < 8; ++j) pm = fmaxf(pm, lds_s[r * 132 + g * 8 + j]);
            lds_red[r * 16 + g] = pm;
        }
        __syncthreads();
        // ---- Phase 2b: merge max, rescale factor ----
        if (tid < 32) {
            float gm = -1e30f;
            for (int g = 0; g < 16; ++g) gm = fmaxf(gm, lds_red[tid * 16 + g]);
            float mn = fmaxf(lds_m[tid], gm);
            lds_f[tid] = __expf(lds_m[tid] - mn);
            lds_m[tid] = mn;
        }
        __syncthreads();
        // ---- Phase 2c: P = exp(S - m), partial row sums ----
        {
            int r = tid & 31, g = tid >> 5;
            float mn = lds_m[r], ps = 0.0f;
            for (int j = 0; j < 8; ++j) {
                float p = __expf(lds_s[r * 132 + g * 8 + j] - mn);
                lds_p[r * 136 + g * 8 + j] = (half_t)p;
                ps += p;
            }
            lds_red[r * 16 + g] = ps;
        }
        __syncthreads();
        // ---- Phase 2d: running denominator ----
        if (tid < 32) {
            float s = 0.0f;
            for (int g = 0; g < 16; ++g) s += lds_red[tid * 16 + g];
            lds_l[tid] = lds_l[tid] * lds_f[tid] + s;
        }
        // ---- Phase 3: rescale O, accumulate P @ V ----
        float f0[8];
        for (int v = 0; v < 8; ++v) f0[v] = lds_f[rg * 16 + v + hi8];
        for (int s = 0; s < 8; ++s)
            for (int v = 0; v < 8; ++v) oacc[s][v] *= f0[v];
        for (int kvc = 0; kvc < 128; kvc += 32) {
            v16h a;
            const half_t* ab = &lds_p[arow * 136 + kvc + off16];
            *(uint4*)&a       = *(const uint4*)ab;
            *((uint4*)&a + 1) = *(const uint4*)(ab + 8);
            for (int s = 0; s < 8; ++s) {
                const half_t* bb =
                    &VT16[((size_t)b * Dm + d0 + s * 16 + col) * LKVn + kv0 + kvc + off16];
                v16h bf;
                *(uint4*)&bf       = *(const uint4*)bb;
                *((uint4*)&bf + 1) = *(const uint4*)(bb + 8);
                oacc[s] = __builtin_amdgcn_wmma_f32_16x16x32_f16(
                              false, a, false, bf, (short)0, oacc[s], false, false);
            }
        }
    }
    __syncthreads();

    // ---- Epilogue: O / l ----
    float rl[8];
    for (int v = 0; v < 8; ++v) rl[v] = 1.0f / lds_l[rg * 16 + v + hi8];
    for (int s = 0; s < 8; ++s) {
        int n = d0 + s * 16 + col;
        for (int v = 0; v < 8; ++v) {
            int m = q0 + rg * 16 + v + hi8;
            Out[((size_t)b * LQn + m) * Dm + n] = oacc[s][v] * rl[v];
        }
    }
}

extern "C" void kernel_launch(void* const* d_in, const int* in_sizes, int n_in,
                              void* d_out, int out_size, void* d_ws, size_t ws_size,
                              hipStream_t stream) {
    const float* query = (const float*)d_in[0];
    const float* key   = (const float*)d_in[1];
    const float* value = (const float*)d_in[2];
    const float* Wq    = (const float*)d_in[3];
    const float* bq    = (const float*)d_in[4];
    const float* Wk    = (const float*)d_in[5];
    const float* bk    = (const float*)d_in[6];
    const int*   scale = (const int*)d_in[7];
    float* out = (float*)d_out;

    char* ws = (char*)d_ws;
    half_t* q16  = (half_t*)ws; ws += (size_t)Bsz * LQn  * Dm * sizeof(half_t);
    half_t* k16  = (half_t*)ws; ws += (size_t)Bsz * LKVn * Dm * sizeof(half_t);
    half_t* vt16 = (half_t*)ws; ws += (size_t)Bsz * Dm * LKVn * sizeof(half_t);
    half_t* wq16 = (half_t*)ws; ws += (size_t)Dm * Dm * sizeof(half_t);
    half_t* wk16 = (half_t*)ws; ws += (size_t)Dm * Dm * sizeof(half_t);

    cvt_f32_f16<<<(Dm * Dm) / 256, 256, 0, stream>>>(Wq, wq16, Dm * Dm);
    cvt_f32_f16<<<(Dm * Dm) / 256, 256, 0, stream>>>(Wk, wk16, Dm * Dm);
    transpose_v<<<dim3(LKVn / 32, Dm / 32, Bsz), dim3(32, 32), 0, stream>>>(value, vt16);
    proj_gemm<<<dim3(Dm / 128, (Bsz * LQn)  / 128), 256, 0, stream>>>(query, wq16, bq, q16);
    proj_gemm<<<dim3(Dm / 128, (Bsz * LKVn) / 128), 256, 0, stream>>>(key,   wk16, bk, k16);
    attn_fused<<<dim3(LQn / 32, Bsz), 512, 0, stream>>>(q16, k16, vt16, scale, out);
}
